// GVAE_59493886984707
// MI455X (gfx1250) — matrix operand order repaired and checked by reading, not compiled, not run on hardware
//
#include <hip/hip_runtime.h>
#include <hip/hip_bf16.h>
#include <math.h>

// ---------------------------------------------------------------------------
// Types for WMMA
// ---------------------------------------------------------------------------
typedef __attribute__((ext_vector_type(16))) _Float16 v16h;
typedef __attribute__((ext_vector_type(8)))  _Float16 v8h;
typedef __attribute__((ext_vector_type(8)))  float    v8f;

// Problem constants
constexpr int N_    = 65536;
constexpr int E_    = 1048576;
constexpr int B_    = 512;
constexpr int FEAT_ = 32;
constexpr int ENC_  = 64;
constexpr int H_    = 4;
constexpr int LAT_  = 128;
constexpr int DEC_  = 512;
constexpr int AO_   = 1280;   // MAX_ATOMS*(ATOM_TYPES+1)
constexpr int EO_   = 8128;   // MAX_ATOMS*(MAX_ATOMS-1)/2
constexpr int QKVS_ = 832;    // 3*H*ENC + ENC (q|k|v|skip fused columns)
constexpr int CHUNK_= 131072; // edges per e-embedding chunk (E_/CHUNK_ = 8)

// ---------------------------------------------------------------------------
// Generic f16 WMMA GEMM:  C[M,Ncols] = A[M,K](row-major f16) * B (given as
// BT[Ncols,K], i.e. K-major / transposed f16) + bias, optional ReLU.
// One wave per 16x16 output tile; K stepped by 32 per v_wmma_f32_16x16x32_f16.
// A-operand lane layout used here (contiguous k, not the ISA table order; 16-bit A 16x32):
//   lane<16: M=lane, halves = K{0..15}; lane>=16: K{16..31}
// B-operand (32x16): lane<16: N=lane, halves = K{0..15}; lane>=16: K{16..31}
// C/D f32: lane<16 -> (m=r, n=lane); lane>=16 -> (m=8+r, n=lane-16)
// ---------------------------------------------------------------------------
__global__ __launch_bounds__(256) void wmma_gemm(
    const _Float16* __restrict__ A, int lda,
    const _Float16* __restrict__ BT, int ldb,
    const float* __restrict__ bias,
    float* __restrict__ Cf, int ldc,
    _Float16* __restrict__ Ch, int ldch,
    int M, int Ncols, int K, int relu)
{
  int wave = blockIdx.x * (blockDim.x >> 5) + (threadIdx.x >> 5);
  int lane = threadIdx.x & 31;
  int tilesN = Ncols >> 4;
  int totalTiles = (M >> 4) * tilesN;
  if (wave >= totalTiles) return;
  int mt = wave / tilesN;
  int nt = wave - mt * tilesN;
  int lhalf = lane & 15;
  int hi    = lane >> 4;

  const _Float16* ap = A  + (size_t)(mt * 16 + lhalf) * lda + hi * 16;
  const _Float16* bp = BT + (size_t)(nt * 16 + lhalf) * ldb + hi * 16;

  v8f acc = {};
  for (int k0 = 0; k0 < K; k0 += 32) {
    __builtin_prefetch((const void*)(ap + k0 + 64), 0, 0);
    v8h  alo = *(const v8h*)(ap + k0);
    v8h  ahi = *(const v8h*)(ap + k0 + 8);
    v16h b   = *(const v16h*)(bp + k0);
    v16h a;
#pragma unroll
    for (int j = 0; j < 8; ++j) { a[j] = alo[j]; a[j + 8] = ahi[j]; }
    acc = __builtin_amdgcn_wmma_f32_16x16x32_f16(
        false, a, false, b, (short)0, acc, false, false);
  }

  int   cn = nt * 16 + lhalf;
  int   mb = mt * 16 + hi * 8;
  float bv = bias ? bias[cn] : 0.0f;
#pragma unroll
  for (int r = 0; r < 8; ++r) {
    float xv = acc[r] + bv;
    if (relu) xv = fmaxf(xv, 0.0f);
    if (Cf) Cf[(size_t)(mb + r) * ldc + cn] = xv;
    if (Ch) Ch[(size_t)(mb + r) * ldch + cn] = (_Float16)xv;
  }
}

// ---------------------------------------------------------------------------
// Small utility kernels
// ---------------------------------------------------------------------------
__global__ void cvt_f32_f16(const float* __restrict__ s, _Float16* __restrict__ d, int n) {
  int t = blockIdx.x * blockDim.x + threadIdx.x;
  if (t < n) d[t] = (_Float16)s[t];
}

__global__ void fill_f32(float* __restrict__ p, float v, int n) {
  int t = blockIdx.x * blockDim.x + threadIdx.x;
  if (t < n) p[t] = v;
}

// W [K,Ncols] f32 -> WT [Ncols,K] f16
__global__ void transpose_w(const float* __restrict__ W, _Float16* __restrict__ WT,
                            int K, int Ncols) {
  int t = blockIdx.x * blockDim.x + threadIdx.x;
  if (t >= K * Ncols) return;
  int nn = t / K, kk = t - nn * K;
  WT[t] = (_Float16)W[(size_t)kk * Ncols + nn];
}

// Fuse Wq|Wk|Wv|Ws into K-major f16 [832,K]
__global__ void pack_qkvs_w(const float* __restrict__ Wq, const float* __restrict__ Wk,
                            const float* __restrict__ Wv, const float* __restrict__ Ws,
                            _Float16* __restrict__ WT, int K) {
  int t = blockIdx.x * blockDim.x + threadIdx.x;
  if (t >= QKVS_ * K) return;
  int c = t / K, k = t - c * K;
  float v;
  if      (c < 256) v = Wq[(size_t)k * 256 + c];
  else if (c < 512) v = Wk[(size_t)k * 256 + (c - 256)];
  else if (c < 768) v = Wv[(size_t)k * 256 + (c - 512)];
  else              v = Ws[(size_t)k * 64  + (c - 768)];
  WT[t] = (_Float16)v;
}

__global__ void pack_qkvs_b(const float* __restrict__ bq, const float* __restrict__ bk,
                            const float* __restrict__ bv, const float* __restrict__ bs,
                            float* __restrict__ bf) {
  int c = blockIdx.x * blockDim.x + threadIdx.x;
  if (c >= QKVS_) return;
  float v;
  if      (c < 256) v = bq[c];
  else if (c < 512) v = bk[c - 256];
  else if (c < 768) v = bv[c - 512];
  else              v = bs[c - 768];
  bf[c] = v;
}

__device__ __forceinline__ float atomicMaxF(float* addr, float val) {
  if (val >= 0.0f)
    return __int_as_float(atomicMax((int*)addr, __float_as_int(val)));
  else
    return __uint_as_float(atomicMin((unsigned int*)addr, __float_as_uint(val)));
}

__device__ __forceinline__ float sigm(float x) { return 1.0f / (1.0f + __expf(-x)); }

// ---------------------------------------------------------------------------
// Edge attention pass 1: logits[e,h] = (q[dst]·(k[src]+e_emb))/8, seg-max to m
// ---------------------------------------------------------------------------
__global__ void edge_logits(const _Float16* __restrict__ qkvs,
                            const _Float16* __restrict__ e16,
                            const int* __restrict__ ei,
                            int e0, float* __restrict__ logits,
                            float* __restrict__ m) {
  int t = blockIdx.x * blockDim.x + threadIdx.x;
  if (t >= CHUNK_ * H_) return;
  int le = t >> 2, h = t & 3;
  int e = e0 + le;
  int src = ei[e], dst = ei[E_ + e];
  const _Float16* q  = qkvs + (size_t)dst * QKVS_ + h * 64;
  const _Float16* k  = qkvs + (size_t)src * QKVS_ + 256 + h * 64;
  const _Float16* em = e16 + (size_t)le * 256 + h * 64;
  float acc = 0.0f;
#pragma unroll 8
  for (int d = 0; d < 64; ++d)
    acc += (float)q[d] * ((float)k[d] + (float)em[d]);
  float lg = acc * 0.125f;  // 1/sqrt(ENC)
  logits[(size_t)e * H_ + h] = lg;
  atomicMaxF(&m[(size_t)dst * H_ + h], lg);
}

// ---------------------------------------------------------------------------
// Edge attention pass 2: ex = exp(l - m[dst]); den += ex; num += ex*(v[src]+e)
// ---------------------------------------------------------------------------
__global__ void edge_accum(const _Float16* __restrict__ qkvs,
                           const _Float16* __restrict__ e16,
                           const int* __restrict__ ei,
                           int e0, const float* __restrict__ logits,
                           const float* __restrict__ m,
                           float* __restrict__ den, float* __restrict__ num) {
  int t = blockIdx.x * blockDim.x + threadIdx.x;
  if (t >= CHUNK_ * H_) return;
  int le = t >> 2, h = t & 3;
  int e = e0 + le;
  int src = ei[e], dst = ei[E_ + e];
  float ex = __expf(logits[(size_t)e * H_ + h] - m[(size_t)dst * H_ + h]);
  atomicAdd(&den[(size_t)dst * H_ + h], ex);
  const _Float16* v  = qkvs + (size_t)src * QKVS_ + 512 + h * 64;
  const _Float16* em = e16 + (size_t)le * 256 + h * 64;
  float* np = num + ((size_t)dst * H_ + h) * 64;
#pragma unroll 8
  for (int d = 0; d < 64; ++d)
    atomicAdd(&np[d], ex * ((float)v[d] + (float)em[d]));
}

// ---------------------------------------------------------------------------
// Node finalize: out = mean_h(num/den); beta gate with skip; ReLU
// ---------------------------------------------------------------------------
__global__ void node_final(const _Float16* __restrict__ qkvs,
                           const float* __restrict__ num,
                           const float* __restrict__ den,
                           const float* __restrict__ Wbeta,
                           float* __restrict__ hout) {
  int i = blockIdx.x * blockDim.x + threadIdx.x;
  if (i >= N_) return;
  float o[64];
#pragma unroll
  for (int d = 0; d < 64; ++d) o[d] = 0.0f;
  for (int h = 0; h < H_; ++h) {
    float dh = den[(size_t)i * H_ + h];
    if (dh > 0.0f) {
      float inv = 1.0f / dh;
      const float* np = num + ((size_t)i * H_ + h) * 64;
      for (int d = 0; d < 64; ++d) o[d] += np[d] * inv;
    }
  }
  const _Float16* sp = qkvs + (size_t)i * QKVS_ + 768;
  float bacc = 0.0f;
  for (int d = 0; d < 64; ++d) {
    float od = o[d] * 0.25f;   // mean over heads
    o[d] = od;
    float sk = (float)sp[d];
    bacc += od * Wbeta[d] + sk * Wbeta[64 + d] + (od - sk) * Wbeta[128 + d];
  }
  float beta = sigm(bacc);
  for (int d = 0; d < 64; ++d) {
    float sk = (float)sp[d];
    float v = beta * sk + (1.0f - beta) * o[d];
    hout[(size_t)i * 64 + d] = fmaxf(v, 0.0f);
  }
}

// ---------------------------------------------------------------------------
// BatchNorm (training mode, biased var) over node dimension
// ---------------------------------------------------------------------------
__global__ void bn_stats(const float* __restrict__ h, float* __restrict__ sum,
                         float* __restrict__ ss, int rowsPerBlock) {
  int f = threadIdx.x;           // 64 threads
  int r0 = blockIdx.x * rowsPerBlock;
  float s = 0.0f, q = 0.0f;
  for (int r = 0; r < rowsPerBlock; ++r) {
    float v = h[(size_t)(r0 + r) * 64 + f];
    s += v; q += v * v;
  }
  atomicAdd(&sum[f], s);
  atomicAdd(&ss[f], q);
}

__global__ void bn_apply(const float* __restrict__ h, const float* __restrict__ sum,
                         const float* __restrict__ ss, const float* __restrict__ g,
                         const float* __restrict__ b, _Float16* __restrict__ h16) {
  int t = blockIdx.x * blockDim.x + threadIdx.x;
  if (t >= N_ * 64) return;
  int f = t & 63;
  float mean = sum[f] * (1.0f / N_);
  float var  = ss[f] * (1.0f / N_) - mean * mean;
  float y = g[f] * (h[t] - mean) * rsqrtf(var + 1e-5f) + b[f];
  h16[t] = (_Float16)y;
}

// ---------------------------------------------------------------------------
// Set2Set kernels
// ---------------------------------------------------------------------------
__global__ void s2s_gates(const float* __restrict__ qstar, const float* __restrict__ hs,
                          const float* __restrict__ Wi, const float* __restrict__ bi,
                          const float* __restrict__ Wh, const float* __restrict__ bh,
                          float* __restrict__ gates) {
  int t = blockIdx.x * blockDim.x + threadIdx.x;
  if (t >= B_ * 256) return;
  int b = t >> 8, j = t & 255;
  float g = bi[j] + bh[j];
  const float* qs = qstar + (size_t)b * 128;
  const float* hb = hs + (size_t)b * 64;
  for (int k = 0; k < 128; ++k) g += qs[k] * Wi[(size_t)k * 256 + j];
  for (int k = 0; k < 64;  ++k) g += hb[k] * Wh[(size_t)k * 256 + j];
  gates[t] = g;
}

__global__ void s2s_cell(const float* __restrict__ gates, float* __restrict__ hs,
                         float* __restrict__ cs) {
  int t = blockIdx.x * blockDim.x + threadIdx.x;
  if (t >= B_ * 64) return;
  int b = t >> 6, d = t & 63;
  const float* g = gates + (size_t)b * 256;
  float gi = g[d], gf = g[64 + d], gg = g[128 + d], go = g[192 + d];
  float c = sigm(gf) * cs[t] + sigm(gi) * tanhf(gg);
  cs[t] = c;
  hs[t] = sigm(go) * tanhf(c);
}

__global__ void s2s_dot(const float* __restrict__ x, const int* __restrict__ batch,
                        const float* __restrict__ hs, float* __restrict__ atte,
                        float* __restrict__ mb) {
  int n = blockIdx.x * blockDim.x + threadIdx.x;
  if (n >= N_) return;
  int b = batch[n];
  const float* xn = x + (size_t)n * 64;
  const float* hb = hs + (size_t)b * 64;
  float e = 0.0f;
  for (int d = 0; d < 64; ++d) e += xn[d] * hb[d];
  atte[n] = e;
  atomicMaxF(&mb[b], e);
}

__global__ void s2s_ex(const float* __restrict__ atte, const float* __restrict__ mb,
                       const int* __restrict__ batch, float* __restrict__ attex,
                       float* __restrict__ denb) {
  int n = blockIdx.x * blockDim.x + threadIdx.x;
  if (n >= N_) return;
  int b = batch[n];
  float ex = __expf(atte[n] - mb[b]);
  attex[n] = ex;
  atomicAdd(&denb[b], ex);
}

__global__ void s2s_racc(const float* __restrict__ attex, const float* __restrict__ denb,
                         const int* __restrict__ batch, const float* __restrict__ x,
                         float* __restrict__ rb) {
  int n = blockIdx.x * blockDim.x + threadIdx.x;
  if (n >= N_) return;
  int b = batch[n];
  float a = attex[n] / denb[b];
  const float* xn = x + (size_t)n * 64;
  float* rbb = rb + (size_t)b * 64;
  for (int d = 0; d < 64; ++d) atomicAdd(&rbb[d], a * xn[d]);
}

__global__ void s2s_qstar(const float* __restrict__ hs, const float* __restrict__ rb,
                          float* __restrict__ qstar) {
  int t = blockIdx.x * blockDim.x + threadIdx.x;
  if (t >= B_ * 64) return;
  int b = t >> 6, d = t & 63;
  qstar[(size_t)b * 128 + d]      = hs[t];
  qstar[(size_t)b * 128 + 64 + d] = rb[t];
}

// z = eps*exp(logvar) + mu  (source uses exp(logvar), not exp(0.5*logvar))
__global__ void reparam(const float* __restrict__ eps, const float* __restrict__ mu,
                        const float* __restrict__ lv, float* __restrict__ z,
                        _Float16* __restrict__ z16) {
  int t = blockIdx.x * blockDim.x + threadIdx.x;
  if (t >= B_ * LAT_) return;
  float v = eps[t] * __expf(lv[t]) + mu[t];
  z[t] = v;
  z16[t] = (_Float16)v;
}

// ---------------------------------------------------------------------------
// Host orchestration
// ---------------------------------------------------------------------------
static inline int blks(long n, int b) { return (int)((n + b - 1) / b); }

extern "C" void kernel_launch(void* const* d_in, const int* in_sizes, int n_in,
                              void* d_out, int out_size, void* d_ws, size_t ws_size,
                              hipStream_t stream) {
  (void)in_sizes; (void)n_in; (void)out_size; (void)ws_size;
  int idx = 0;
  const float* x          = (const float*)d_in[idx++];
  const float* edge_attr  = (const float*)d_in[idx++];
  const int*   edge_index = (const int*)d_in[idx++];
  const int*   batch_ix   = (const int*)d_in[idx++];
  const float* eps        = (const float*)d_in[idx++];
  struct Layer { const float *Wq,*bq,*Wk,*bk,*Wv,*bv,*We,*be,*Ws,*bs,*Wbeta,*bng,*bnb; };
  Layer L[4];
  for (int l = 0; l < 4; ++l) {
    L[l].Wq = (const float*)d_in[idx++]; L[l].bq = (const float*)d_in[idx++];
    L[l].Wk = (const float*)d_in[idx++]; L[l].bk = (const float*)d_in[idx++];
    L[l].Wv = (const float*)d_in[idx++]; L[l].bv = (const float*)d_in[idx++];
    L[l].We = (const float*)d_in[idx++]; L[l].be = (const float*)d_in[idx++];
    L[l].Ws = (const float*)d_in[idx++]; L[l].bs = (const float*)d_in[idx++];
    L[l].Wbeta = (const float*)d_in[idx++];
    if (l < 3) { L[l].bng = (const float*)d_in[idx++]; L[l].bnb = (const float*)d_in[idx++]; }
    else       { L[l].bng = nullptr; L[l].bnb = nullptr; }
  }
  const float* lstm_Wi = (const float*)d_in[idx++];
  const float* lstm_bi = (const float*)d_in[idx++];
  const float* lstm_Wh = (const float*)d_in[idx++];
  const float* lstm_bh = (const float*)d_in[idx++];
  const float* Wmu = (const float*)d_in[idx++]; const float* bmu = (const float*)d_in[idx++];
  const float* Wlv = (const float*)d_in[idx++]; const float* blv = (const float*)d_in[idx++];
  const float* Wl1 = (const float*)d_in[idx++]; const float* bl1 = (const float*)d_in[idx++];
  const float* Wl2 = (const float*)d_in[idx++]; const float* bl2 = (const float*)d_in[idx++];
  const float* Watom = (const float*)d_in[idx++]; const float* batom = (const float*)d_in[idx++];
  const float* Wedge = (const float*)d_in[idx++]; const float* bedge = (const float*)d_in[idx++];

  // Workspace carve-out
  size_t off = 0;
  auto alloc = [&](size_t bytes) -> char* {
    char* p = (char*)d_ws + off;
    off += (bytes + 255) & ~(size_t)255;
    return p;
  };
  _Float16* h16    = (_Float16*)alloc((size_t)N_ * 64 * 2);
  _Float16* qkvs16 = (_Float16*)alloc((size_t)N_ * QKVS_ * 2);
  float*    logits = (float*)alloc((size_t)E_ * H_ * 4);
  float*    mmax   = (float*)alloc((size_t)N_ * H_ * 4);
  float*    den    = (float*)alloc((size_t)N_ * H_ * 4);
  float*    num    = (float*)alloc((size_t)N_ * H_ * 64 * 4);
  _Float16* e16    = (_Float16*)alloc((size_t)CHUNK_ * 256 * 2);
  _Float16* ea16   = (_Float16*)alloc((size_t)E_ * FEAT_ * 2);
  float*    hnext  = (float*)alloc((size_t)N_ * 64 * 4);
  float*    bnsum  = (float*)alloc(64 * 4);
  float*    bnss   = (float*)alloc(64 * 4);
  _Float16* wq16[4]; float* bq32[4]; _Float16* we16[4];
  for (int l = 0; l < 4; ++l) {
    wq16[l] = (_Float16*)alloc((size_t)QKVS_ * 64 * 2);
    bq32[l] = (float*)alloc(QKVS_ * 4);
    we16[l] = (_Float16*)alloc((size_t)256 * 32 * 2);
  }
  float* hs    = (float*)alloc((size_t)B_ * 64 * 4);
  float* cs    = (float*)alloc((size_t)B_ * 64 * 4);
  float* qstar = (float*)alloc((size_t)B_ * 128 * 4);
  float* gates = (float*)alloc((size_t)B_ * 256 * 4);
  float* atte  = (float*)alloc((size_t)N_ * 4);
  float* attex = (float*)alloc((size_t)N_ * 4);
  float* mb    = (float*)alloc((size_t)B_ * 4);
  float* denb  = (float*)alloc((size_t)B_ * 4);
  float* rb    = (float*)alloc((size_t)B_ * 64 * 4);
  _Float16* pooled16 = (_Float16*)alloc((size_t)B_ * 128 * 2);
  float*    zbuf  = (float*)alloc((size_t)B_ * LAT_ * 4);
  _Float16* z16   = (_Float16*)alloc((size_t)B_ * LAT_ * 2);
  _Float16* d1_16 = (_Float16*)alloc((size_t)B_ * DEC_ * 2);
  _Float16* d2_16 = (_Float16*)alloc((size_t)B_ * DEC_ * 2);
  _Float16* WmuT = (_Float16*)alloc((size_t)LAT_ * 128 * 2);
  _Float16* WlvT = (_Float16*)alloc((size_t)LAT_ * 128 * 2);
  _Float16* Wl1T = (_Float16*)alloc((size_t)DEC_ * LAT_ * 2);
  _Float16* Wl2T = (_Float16*)alloc((size_t)DEC_ * DEC_ * 2);
  _Float16* WatT = (_Float16*)alloc((size_t)AO_ * DEC_ * 2);
  _Float16* WedT = (_Float16*)alloc((size_t)EO_ * DEC_ * 2);

  // Output layout (f32): [triu | node | mu | logvar]
  float* out_triu = (float*)d_out;
  float* out_node = out_triu + (size_t)B_ * EO_;
  float* out_mu   = out_node + (size_t)B_ * AO_;
  float* out_lv   = out_mu   + (size_t)B_ * LAT_;

  const int TB = 256;

  // ---- weight prep (f16, K-major) ----
  cvt_f32_f16<<<blks((long)N_ * FEAT_, TB), TB, 0, stream>>>(x, h16, N_ * FEAT_);
  cvt_f32_f16<<<blks((long)E_ * FEAT_, TB), TB, 0, stream>>>(edge_attr, ea16, E_ * FEAT_);
  for (int l = 0; l < 4; ++l) {
    int K = (l == 0) ? FEAT_ : ENC_;
    pack_qkvs_w<<<blks((long)QKVS_ * K, TB), TB, 0, stream>>>(L[l].Wq, L[l].Wk, L[l].Wv, L[l].Ws, wq16[l], K);
    pack_qkvs_b<<<blks(QKVS_, TB), TB, 0, stream>>>(L[l].bq, L[l].bk, L[l].bv, L[l].bs, bq32[l]);
    transpose_w<<<blks((long)32 * 256, TB), TB, 0, stream>>>(L[l].We, we16[l], 32, 256);
  }
  transpose_w<<<blks((long)LAT_ * 128, TB), TB, 0, stream>>>(Wmu, WmuT, 128, 128);
  transpose_w<<<blks((long)LAT_ * 128, TB), TB, 0, stream>>>(Wlv, WlvT, 128, 128);
  transpose_w<<<blks((long)LAT_ * DEC_, TB), TB, 0, stream>>>(Wl1, Wl1T, LAT_, DEC_);
  transpose_w<<<blks((long)DEC_ * DEC_, TB), TB, 0, stream>>>(Wl2, Wl2T, DEC_, DEC_);
  transpose_w<<<blks((long)DEC_ * AO_, TB), TB, 0, stream>>>(Watom, WatT, DEC_, AO_);
  transpose_w<<<blks((long)DEC_ * EO_, TB), TB, 0, stream>>>(Wedge, WedT, DEC_, EO_);

  // ---- 4 TransformerConv layers ----
  for (int l = 0; l < 4; ++l) {
    int K = (l == 0) ? FEAT_ : ENC_;
    // fused node GEMM: qkvs16[N,832] = h16[N,K] x W + b
    {
      long tiles = (long)(N_ / 16) * (QKVS_ / 16);
      wmma_gemm<<<blks(tiles, 8), 256, 0, stream>>>(
          h16, K, wq16[l], K, bq32[l], nullptr, 0, qkvs16, QKVS_,
          N_, QKVS_, K, 0);
    }
    fill_f32<<<blks((long)N_ * H_, TB), TB, 0, stream>>>(mmax, -INFINITY, N_ * H_);
    hipMemsetAsync(den, 0, (size_t)N_ * H_ * 4, stream);
    hipMemsetAsync(num, 0, (size_t)N_ * H_ * 64 * 4, stream);

    long etiles = (long)(CHUNK_ / 16) * (256 / 16);
    // pass 1: logits + segment max
    for (int c = 0; c < E_ / CHUNK_; ++c) {
      int e0 = c * CHUNK_;
      wmma_gemm<<<blks(etiles, 8), 256, 0, stream>>>(
          ea16 + (size_t)e0 * FEAT_, FEAT_, we16[l], FEAT_, L[l].be,
          nullptr, 0, e16, 256, CHUNK_, 256, FEAT_, 0);
      edge_logits<<<blks((long)CHUNK_ * H_, TB), TB, 0, stream>>>(
          qkvs16, e16, edge_index, e0, logits, mmax);
    }
    // pass 2: exp/sum + weighted scatter
    for (int c = 0; c < E_ / CHUNK_; ++c) {
      int e0 = c * CHUNK_;
      wmma_gemm<<<blks(etiles, 8), 256, 0, stream>>>(
          ea16 + (size_t)e0 * FEAT_, FEAT_, we16[l], FEAT_, L[l].be,
          nullptr, 0, e16, 256, CHUNK_, 256, FEAT_, 0);
      edge_accum<<<blks((long)CHUNK_ * H_, TB), TB, 0, stream>>>(
          qkvs16, e16, edge_index, e0, logits, mmax, den, num);
    }
    node_final<<<blks(N_, TB), TB, 0, stream>>>(qkvs16, num, den, L[l].Wbeta, hnext);
    if (l < 3) {
      hipMemsetAsync(bnsum, 0, 64 * 4, stream);
      hipMemsetAsync(bnss, 0, 64 * 4, stream);
      bn_stats<<<256, 64, 0, stream>>>(hnext, bnsum, bnss, N_ / 256);
      bn_apply<<<blks((long)N_ * 64, TB), TB, 0, stream>>>(
          hnext, bnsum, bnss, L[l].bng, L[l].bnb, h16);
    }
  }

  // ---- Set2Set (4 processing steps) ----
  hipMemsetAsync(hs, 0, (size_t)B_ * 64 * 4, stream);
  hipMemsetAsync(cs, 0, (size_t)B_ * 64 * 4, stream);
  hipMemsetAsync(qstar, 0, (size_t)B_ * 128 * 4, stream);
  for (int s = 0; s < 4; ++s) {
    s2s_gates<<<blks((long)B_ * 256, TB), TB, 0, stream>>>(
        qstar, hs, lstm_Wi, lstm_bi, lstm_Wh, lstm_bh, gates);
    s2s_cell<<<blks((long)B_ * 64, TB), TB, 0, stream>>>(gates, hs, cs);
    fill_f32<<<blks(B_, TB), TB, 0, stream>>>(mb, -INFINITY, B_);
    hipMemsetAsync(denb, 0, (size_t)B_ * 4, stream);
    hipMemsetAsync(rb, 0, (size_t)B_ * 64 * 4, stream);
    s2s_dot<<<blks(N_, TB), TB, 0, stream>>>(hnext, batch_ix, hs, atte, mb);
    s2s_ex<<<blks(N_, TB), TB, 0, stream>>>(atte, mb, batch_ix, attex, denb);
    s2s_racc<<<blks(N_, TB), TB, 0, stream>>>(attex, denb, batch_ix, hnext, rb);
    s2s_qstar<<<blks((long)B_ * 64, TB), TB, 0, stream>>>(hs, rb, qstar);
  }
  cvt_f32_f16<<<blks((long)B_ * 128, TB), TB, 0, stream>>>(qstar, pooled16, B_ * 128);

  // ---- mu / logvar (WMMA) + reparameterize ----
  {
    long tiles = (long)(B_ / 16) * (LAT_ / 16);
    wmma_gemm<<<blks(tiles, 8), 256, 0, stream>>>(
        pooled16, 128, WmuT, 128, bmu, out_mu, LAT_, nullptr, 0, B_, LAT_, 128, 0);
    wmma_gemm<<<blks(tiles, 8), 256, 0, stream>>>(
        pooled16, 128, WlvT, 128, blv, out_lv, LAT_, nullptr, 0, B_, LAT_, 128, 0);
  }
  reparam<<<blks((long)B_ * LAT_, TB), TB, 0, stream>>>(eps, out_mu, out_lv, zbuf, z16);

  // ---- decoder (WMMA) ----
  {
    long t1 = (long)(B_ / 16) * (DEC_ / 16);
    wmma_gemm<<<blks(t1, 8), 256, 0, stream>>>(
        z16, LAT_, Wl1T, LAT_, bl1, nullptr, 0, d1_16, DEC_, B_, DEC_, LAT_, 1);
    wmma_gemm<<<blks(t1, 8), 256, 0, stream>>>(
        d1_16, DEC_, Wl2T, DEC_, bl2, nullptr, 0, d2_16, DEC_, B_, DEC_, DEC_, 1);
    long ta = (long)(B_ / 16) * (AO_ / 16);
    wmma_gemm<<<blks(ta, 8), 256, 0, stream>>>(
        d2_16, DEC_, WatT, DEC_, batom, out_node, AO_, nullptr, 0, B_, AO_, DEC_, 0);
    long te = (long)(B_ / 16) * (EO_ / 16);
    wmma_gemm<<<blks(te, 8), 256, 0, stream>>>(
        d2_16, DEC_, WedT, DEC_, bedge, out_triu, EO_, nullptr, 0, B_, EO_, DEC_, 0);
  }
}
